// E3GNN_24154896073540
// MI455X (gfx1250) — matrix operand order repaired and checked by reading, not compiled, not run on hardware
//
#include <hip/hip_runtime.h>
#include <stdint.h>

// ---------------------------------------------------------------------------
// E3GNN on MI455X (gfx1250): tensor product as bf16 WMMA GEMMs over 16-edge
// tiles, f32 accumulate, L2-resident gather/scatter with f32 atomics.
// Round 2: keep per-lane VGPR residency < 256 (no MSB banks, no spills) by
// staging the v1 features in a per-wave LDS slab read as broadcast b64s.
// ---------------------------------------------------------------------------

typedef __attribute__((ext_vector_type(16))) __bf16 v16bf;
typedef __attribute__((ext_vector_type(8)))  float  v8f;
typedef __attribute__((ext_vector_type(8)))  unsigned int v8u;

static constexpr float INV_S3      = 0.5773502691896258f;   // 1/sqrt(3)
static constexpr float INV_S2      = 0.7071067811865476f;   // 1/sqrt(2)
static constexpr float NORM_S      = 0.04419417382415922f;  // 1/sqrt(2*16*16)
static constexpr float NORM_V      = 0.03608439182435161f;  // 1/sqrt(3*16*16)
static constexpr float INV_LIN_IN  = 0.35355339059327373f;  // 1/sqrt(8)
static constexpr float INV_LIN_OUT = 0.25f;                 // 1/sqrt(16)

// Node feature layout: [N][64] f32 = s[0..15], v0[0..15], v1[0..15], v2[0..15]

__device__ __forceinline__ uint32_t f2bf(float f) {
    union { float f; uint32_t u; } x; x.f = f;
    uint32_t u = x.u;
    return ((u + 0x7FFFu + ((u >> 16) & 1u)) >> 16) & 0xFFFFu; // RNE
}

// ---------------------------------------------------------------------------
// Weight prep: scale by CG/fan-in norms, convert to bf16, pre-swizzle into
// WMMA B-fragment lane order.  Per layer: 40 K-chunks of 32 (16 for the
// scalar-output GEMM, 24 for the shared vector-output GEMM), each chunk is
// 32 lanes x 8 dwords (two bf16 per dword) = 1 KB.  B element mapping:
//   lane L, vgpr r, half h, s = 2r + h  ->  K = (s/8)*16 + (L/16)*8 + s%8,  N = L%16
// ---------------------------------------------------------------------------
__global__ void k_prep(const float* __restrict__ tp_w, uint32_t* __restrict__ wfrag) {
    int idx = blockIdx.x * blockDim.x + threadIdx.x;
    if (idx >= 3 * 40 * 256) return;
    int l    = idx / (40 * 256);
    int rem  = idx % (40 * 256);
    int c    = rem / 256;
    int t8   = rem % 256;
    int lane = t8 / 8;
    int r    = t8 % 8;
    int n    = lane & 15;
    uint32_t out = 0;
    #pragma unroll
    for (int h = 0; h < 2; ++h) {
        int s = 2 * r + h;
        int Klocal = (s & 8) * 2 + ((lane >> 4) << 3) + (s & 7);  // 0..31 within chunk
        int p; int row; float scale;
        if (c < 16) {                       // scalar-output GEMM: [W0 ; W3]
            row = c * 32 + Klocal;          // 0..511
            if (row < 256) { p = 0; scale = NORM_S; }
            else           { p = 3; scale = NORM_S * INV_S3; row -= 256; }
        } else {                            // vector-output GEMM: [W1 ; W2 ; W4]
            row = (c - 16) * 32 + Klocal;   // 0..767
            if (row < 256)      { p = 1; scale = NORM_V; }
            else if (row < 512) { p = 2; scale = NORM_V;          row -= 256; }
            else                { p = 4; scale = NORM_V * INV_S2; row -= 512; }
        }
        int u = row >> 4, v = row & 15;
        float w = tp_w[(((size_t)(l * 5 + p) * 16 + u) * 16 + v) * 16 + n] * scale;
        out |= f2bf(w) << (16 * h);
    }
    wfrag[idx] = out;
}

// ---------------------------------------------------------------------------
// Input irreps-Linear; also zeroes the scatter accumulator for layer 0.
// ---------------------------------------------------------------------------
__global__ void k_input(const float* __restrict__ x, const float* __restrict__ w_in0,
                        const float* __restrict__ w_in1, float* __restrict__ nodeA,
                        float* __restrict__ nodeB, int N) {
    int nidx = blockIdx.x * blockDim.x + threadIdx.x;
    if (nidx >= N) return;
    const float* xr = x + (size_t)nidx * 32;
    float* oa = nodeA + (size_t)nidx * 64;
    float* ob = nodeB + (size_t)nidx * 64;
    #pragma unroll
    for (int w = 0; w < 16; ++w) {
        float s = 0.f;
        #pragma unroll
        for (int u = 0; u < 8; ++u) s += xr[u] * w_in0[u * 16 + w];
        oa[w] = s * INV_LIN_IN;
        #pragma unroll
        for (int i = 0; i < 3; ++i) {
            float v = 0.f;
            #pragma unroll
            for (int u = 0; u < 8; ++u) v += xr[8 + u * 3 + i] * w_in1[u * 16 + w];
            oa[16 + 16 * i + w] = v * INV_LIN_IN;
        }
    }
    #pragma unroll
    for (int q = 0; q < 64; ++q) ob[q] = 0.f;
}

// ---------------------------------------------------------------------------
// Edge tensor-product + scatter.  One wave per 16-edge tile (grid-stride).
//   A (16x32 bf16): Z outer-product chunks; lane L holds edge m = L%16,
//                   v-half = L/16.  s1 + b-side halves live in VGPRs; the
//                   48 v1 components live in a per-wave LDS slab laid out
//                   (u-pair)-major: float2 at [(i*8+c)*32 + m*2] = v1_i[2c,2c+1]
//                   -> broadcast, bank-conflict-free ds_load_b64 per chunk.
//   B (32x16 bf16): pre-swizzled weight fragments streamed from LDS.
//   C/D (16x16 f32): 4 accumulators; VGPR j holds edge m = j + 8*(L/16),
//                   output column n = L%16.
// ---------------------------------------------------------------------------
__global__ void __launch_bounds__(128) k_edge(const float* __restrict__ nodeA,
                                              float* __restrict__ nodeB,
                                              const int* __restrict__ row,
                                              const int* __restrict__ col,
                                              const uint32_t* __restrict__ wfrag,
                                              int E) {
    __shared__ uint32_t lds_w[40 * 256];          // 40 KB of bf16 B-fragments
    __shared__ float    lds_a[4 * 768];           // 4 waves x (24 pairs x 32) f32
    for (int i = threadIdx.x; i < 40 * 256; i += blockDim.x) lds_w[i] = wfrag[i];
    __syncthreads();

    const int lane = threadIdx.x & 31;
    const int vh   = lane >> 4;                    // v-half owned by this lane
    const int m    = lane & 15;                    // A-row / edge within tile
    const int ws_  = (threadIdx.x >> 5) * 768;     // this wave's slab base
    const int nw   = (gridDim.x * blockDim.x) >> 5;
    int gw = (blockIdx.x * blockDim.x + threadIdx.x) >> 5;
    const int tiles = (E + 15) >> 4;
    const float4* NA = (const float4*)nodeA;

    #pragma clang loop unroll(disable)
    for (int t = gw; t < tiles; t += nw) {
        const int e  = t * 16 + m;
        const bool ok = e < E;
        const int rN = ok ? row[e] : 0;            // garbage rows are never
        const int cN = ok ? col[e] : 0;            // scattered, so just clamp

        // ---- s1 (all 16 u) into VGPRs -------------------------------------
        float a_s[16];
        #pragma unroll
        for (int q = 0; q < 4; ++q) {
            float4 f = NA[rN * 16 + q];
            a_s[4 * q + 0] = f.x; a_s[4 * q + 1] = f.y;
            a_s[4 * q + 2] = f.z; a_s[4 * q + 3] = f.w;
        }
        // ---- this lane's v-half of s2/v2 into VGPRs -----------------------
        float b_s[8], b_v[3][8];
        {
            const int bb = cN * 16 + vh * 2;
            #pragma unroll
            for (int q = 0; q < 2; ++q) {
                float4 f = NA[bb + q];
                b_s[4 * q + 0] = f.x; b_s[4 * q + 1] = f.y;
                b_s[4 * q + 2] = f.z; b_s[4 * q + 3] = f.w;
            }
            #pragma unroll
            for (int i = 0; i < 3; ++i) {
                #pragma unroll
                for (int q = 0; q < 2; ++q) {
                    float4 f = NA[bb + 4 + i * 4 + q];
                    b_v[i][4 * q + 0] = f.x; b_v[i][4 * q + 1] = f.y;
                    b_v[i][4 * q + 2] = f.z; b_v[i][4 * q + 3] = f.w;
                }
            }
        }
        // ---- cooperative v1 staging into the LDS slab ---------------------
        // 12 float4s per edge; lane m takes lin 0..5, lane m+16 takes 6..11.
        #pragma unroll
        for (int k = 0; k < 6; ++k) {
            const int lin = vh * 6 + k;            // 0..11
            const int i = lin >> 2, q = lin & 3;   // component, float4 index
            float4 f = NA[rN * 16 + 4 + i * 4 + q];
            const int fp0 = i * 8 + 2 * q;         // u-pair index
            *(float2*)&lds_a[ws_ + (fp0    ) * 32 + m * 2] = make_float2(f.x, f.y);
            *(float2*)&lds_a[ws_ + (fp0 + 1) * 32 + m * 2] = make_float2(f.z, f.w);
        }
        // drain this wave's DS stores before the slab reads below
        asm volatile("s_wait_dscnt 0" ::: "memory");

        v8f cs  = {0,0,0,0,0,0,0,0};
        v8f cv0 = {0,0,0,0,0,0,0,0};
        v8f cv1 = {0,0,0,0,0,0,0,0};
        v8f cv2 = {0,0,0,0,0,0,0,0};
        v16bf af, bw;

#define LOADW(chunk) { v8u w8 = *(const v8u*)&lds_w[(chunk) * 256 + lane * 8]; \
                       bw = __builtin_bit_cast(v16bf, w8); }
#define LOADA(i, c) (*(const float2*)&lds_a[ws_ + ((i) * 8 + (c)) * 32 + m * 2])
#define WM(acc) acc = __builtin_amdgcn_wmma_f32_16x16x32_bf16( \
                      false, af, false, bw, (short)0, acc, false, false);
#define BUILD_AF(E0, E1) { _Pragma("unroll") \
        for (int j = 0; j < 8; ++j) { af[j] = (__bf16)(E0); af[j + 8] = (__bf16)(E1); } }

        // ---- out_s: path p1 (s1 x s2 . W0') -------------------------------
        #pragma unroll
        for (int c = 0; c < 8; ++c) {
            const int u0 = 2 * c, u1 = 2 * c + 1;
            LOADW(c);
            BUILD_AF(a_s[u0] * b_s[j], a_s[u1] * b_s[j]);
            WM(cs);
        }
        // ---- out_s: path p4 (v1.v2 dot . W3') -----------------------------
        #pragma unroll
        for (int c = 0; c < 8; ++c) {
            LOADW(8 + c);
            const float2 av0 = LOADA(0, c), av1 = LOADA(1, c), av2 = LOADA(2, c);
            BUILD_AF(av0.x * b_v[0][j] + av1.x * b_v[1][j] + av2.x * b_v[2][j],
                     av0.y * b_v[0][j] + av1.y * b_v[1][j] + av2.y * b_v[2][j]);
            WM(cs);
        }
        // ---- out_v: path p2 (s1 x v2_i . W1'), B-frag reused 3x -----------
        #pragma unroll
        for (int c = 0; c < 8; ++c) {
            const int u0 = 2 * c, u1 = 2 * c + 1;
            LOADW(16 + c);
            BUILD_AF(a_s[u0] * b_v[0][j], a_s[u1] * b_v[0][j]); WM(cv0);
            BUILD_AF(a_s[u0] * b_v[1][j], a_s[u1] * b_v[1][j]); WM(cv1);
            BUILD_AF(a_s[u0] * b_v[2][j], a_s[u1] * b_v[2][j]); WM(cv2);
        }
        // ---- out_v: path p3 (v1_i x s2 . W2') -----------------------------
        #pragma unroll
        for (int c = 0; c < 8; ++c) {
            LOADW(24 + c);
            const float2 av0 = LOADA(0, c), av1 = LOADA(1, c), av2 = LOADA(2, c);
            BUILD_AF(av0.x * b_s[j], av0.y * b_s[j]); WM(cv0);
            BUILD_AF(av1.x * b_s[j], av1.y * b_s[j]); WM(cv1);
            BUILD_AF(av2.x * b_s[j], av2.y * b_s[j]); WM(cv2);
        }
        // ---- out_v: path p5 (cross product . W4'), cyc = (1,2),(2,0),(0,1)
        #pragma unroll
        for (int c = 0; c < 8; ++c) {
            LOADW(32 + c);
            const float2 av0 = LOADA(0, c), av1 = LOADA(1, c), av2 = LOADA(2, c);
            BUILD_AF(av1.x * b_v[2][j] - av2.x * b_v[1][j],
                     av1.y * b_v[2][j] - av2.y * b_v[1][j]); WM(cv0);
            BUILD_AF(av2.x * b_v[0][j] - av0.x * b_v[2][j],
                     av2.y * b_v[0][j] - av0.y * b_v[2][j]); WM(cv1);
            BUILD_AF(av0.x * b_v[1][j] - av1.x * b_v[0][j],
                     av0.y * b_v[1][j] - av1.y * b_v[0][j]); WM(cv2);
        }
#undef LOADW
#undef LOADA
#undef WM
#undef BUILD_AF

        // ---- scatter-add to destination (row) nodes -----------------------
        const int n = lane & 15;
        #pragma unroll
        for (int j = 0; j < 8; ++j) {
            const int mm = j + (vh << 3);          // C/D row held in vgpr j
            const int e2 = t * 16 + mm;
            if (e2 < E) {
                float* p = nodeB + (size_t)row[e2] * 64;
                atomicAdd(p + n,        cs[j]);
                atomicAdd(p + 16 + n,   cv0[j]);
                atomicAdd(p + 32 + n,   cv1[j]);
                atomicAdd(p + 48 + n,   cv2[j]);
            }
        }
    }
}

// ---------------------------------------------------------------------------
// Gate: s' = silu(s);  v' = v * sigmoid(s).  Re-zeroes the accumulator.
// ---------------------------------------------------------------------------
__global__ void k_gate(float* __restrict__ nodeB, float* __restrict__ nodeA, int N) {
    int idx = blockIdx.x * blockDim.x + threadIdx.x;
    if (idx >= N * 16) return;
    int nidx = idx >> 4, w = idx & 15;
    float* pb = nodeB + (size_t)nidx * 64;
    float* pa = nodeA + (size_t)nidx * 64;
    float s = pb[w];
    float g = 1.f / (1.f + __expf(-s));
    pa[w]      = s * g;
    pa[16 + w] = pb[16 + w] * g;
    pa[32 + w] = pb[32 + w] * g;
    pa[48 + w] = pb[48 + w] * g;
    pb[w] = 0.f; pb[16 + w] = 0.f; pb[32 + w] = 0.f; pb[48 + w] = 0.f;
}

__global__ void k_zero_out(float* __restrict__ out, int n) {
    int idx = blockIdx.x * blockDim.x + threadIdx.x;
    if (idx < n) out[idx] = 0.f;
}

__global__ void k_readout(const float* __restrict__ nodeA, const float* __restrict__ w_out0,
                          const int* __restrict__ batch, float* __restrict__ out, int N) {
    int nidx = blockIdx.x * blockDim.x + threadIdx.x;
    if (nidx >= N) return;
    const float* s = nodeA + (size_t)nidx * 64;
    float acc = 0.f;
    #pragma unroll
    for (int w = 0; w < 16; ++w) acc += s[w] * w_out0[w];
    atomicAdd(&out[batch[nidx]], acc * INV_LIN_OUT);
}

// ---------------------------------------------------------------------------
extern "C" void kernel_launch(void* const* d_in, const int* in_sizes, int n_in,
                              void* d_out, int out_size, void* d_ws, size_t ws_size,
                              hipStream_t stream) {
    const float* x      = (const float*)d_in[0];
    const int*   row    = (const int*)d_in[1];
    const int*   col    = (const int*)d_in[2];
    const int*   batch  = (const int*)d_in[3];
    const float* w_in0  = (const float*)d_in[4];
    const float* w_in1  = (const float*)d_in[5];
    const float* tp_w   = (const float*)d_in[6];
    const float* w_out0 = (const float*)d_in[7];

    const int E = in_sizes[1];    // 600000
    const int N = in_sizes[3];    // 50000
    const int G = out_size;       // 64
    float* out = (float*)d_out;

    char* ws = (char*)d_ws;
    size_t nb = (((size_t)N * 64 * 4) + 255) & ~(size_t)255;
    float*    nodeA = (float*)ws;
    float*    nodeB = (float*)(ws + nb);
    uint32_t* wfrag = (uint32_t*)(ws + 2 * nb);   // 3 * 40 * 256 u32 = 120 KB

    k_prep<<<(3 * 40 * 256 + 255) / 256, 256, 0, stream>>>(tp_w, wfrag);
    k_input<<<(N + 255) / 256, 256, 0, stream>>>(x, w_in0, w_in1, nodeA, nodeB, N);

    for (int l = 0; l < 3; ++l) {
        k_edge<<<1024, 128, 0, stream>>>(nodeA, nodeB, row, col,
                                         wfrag + (size_t)l * 40 * 256, E);
        k_gate<<<(N * 16 + 255) / 256, 256, 0, stream>>>(nodeB, nodeA, N);
    }

    k_zero_out<<<1, 256, 0, stream>>>(out, G);
    k_readout<<<(N + 255) / 256, 256, 0, stream>>>(nodeA, w_out0, batch, out, N);
}
